// SelfAttention_8400956031432
// MI455X (gfx1250) — compile-verified, matrix operand order checked by reading, not run on hardware
//
#include <hip/hip_runtime.h>
#include <hip/hip_bf16.h>
#include <stdint.h>

typedef __attribute__((ext_vector_type(16))) _Float16 v16h;
typedef __attribute__((ext_vector_type(8)))  _Float16 v8h;
typedef __attribute__((ext_vector_type(8)))  float    v8f;

// Dims fixed by the reference
#define BB 2
#define NN 2048
#define CC 1024
#define HH 16
#define HD 64

#define QK ((size_t)BB * HH * NN * HD)   // elements per q/k/v tensor (4,194,304)

__device__ __forceinline__ v16h ld16(const _Float16* p0, const _Float16* p1) {
    v8h lo = *(const v8h*)p0;
    v8h hi = *(const v8h*)p1;
    return __builtin_shufflevector(lo, hi, 0,1,2,3,4,5,6,7,8,9,10,11,12,13,14,15);
}

// Async 16B copy global -> LDS (tracked by ASYNCcnt). lds = LDS byte offset
// (low 32 bits of the generic pointer, per ISA aperture rules).
__device__ __forceinline__ void async_cp16(uint32_t lds, const _Float16* gptr) {
    asm volatile("global_load_async_to_lds_b128 %0, %1, off"
                 :: "v"(lds), "v"(gptr) : "memory");
}
__device__ __forceinline__ void wait_async0() {
    asm volatile("s_wait_asynccnt 0x0" ::: "memory");
}

// ---------------------------------------------------------------------------
// Kernel 0a: convert x fp32 -> f16 (row-major, unchanged layout)
// ---------------------------------------------------------------------------
__global__ __launch_bounds__(256) void x_convert_kernel(
    const float* __restrict__ x, _Float16* __restrict__ xh) {
    int i = blockIdx.x * 256 + threadIdx.x;      // 4,194,304 elements
    xh[i] = (_Float16)x[i];
}

// ---------------------------------------------------------------------------
// Kernel 0b: convert + transpose W fp32 [k][n] -> Wt f16 [n][k]
// 32x32 LDS tile transpose, coalesced both sides.
// ---------------------------------------------------------------------------
__global__ __launch_bounds__(256) void w_transpose_kernel(
    const float* __restrict__ W, _Float16* __restrict__ Wt) {
    __shared__ float tile[32][33];
    const int n0 = blockIdx.x * 32;   // 0..3071
    const int k0 = blockIdx.y * 32;   // 0..1023
    const int tx = threadIdx.x & 31, ty = threadIdx.x >> 5;  // 32 x 8
#pragma unroll
    for (int i = 0; i < 32; i += 8)
        tile[ty + i][tx] = W[(size_t)(k0 + ty + i) * (3 * CC) + n0 + tx];
    __syncthreads();
#pragma unroll
    for (int i = 0; i < 32; i += 8)
        Wt[(size_t)(n0 + ty + i) * CC + k0 + tx] = (_Float16)tile[tx][ty + i];
}

// ---------------------------------------------------------------------------
// Kernel 1: QKV projection GEMM (M=4096, K=1024, N=3072), f16 WMMA.
// Tiles staged with GLOBAL_LOAD_ASYNC_TO_LDS_B128 (pure f16 copies).
// q,k written head-major [b][h][n][d]; v written TRANSPOSED [b][h][d][n].
// ---------------------------------------------------------------------------
__global__ __launch_bounds__(128) void qkv_gemm_kernel(
    const _Float16* __restrict__ xh, const _Float16* __restrict__ Wt,
    _Float16* __restrict__ qkv) {
    __shared__ alignas(16) _Float16 As[64 * 40];   // 64 m-rows x 32 K (+8 pad)
    __shared__ alignas(16) _Float16 Bs[64 * 40];   // 64 n-rows x 32 K (+8 pad)

    const int tid  = threadIdx.x;
    const int wv   = tid >> 5;
    const int lane = tid & 31;
    const int hf   = lane >> 4;
    const int lq   = lane & 15;

    const int ntile = blockIdx.x;      // 0..47
    const int mtile = blockIdx.y;      // 0..63
    const int gm0 = mtile * 64;
    const int gn0 = ntile * 64;

    v8f acc[4];
#pragma unroll
    for (int c = 0; c < 4; ++c) acc[c] = {};

    for (int kk = 0; kk < 32; ++kk) {
        const int k0 = kk * 32;
        __syncthreads();   // previous iteration's readers done
        // Async-stage A (xh rows) and B (Wt rows): 256 chunks of 16B each.
#pragma unroll
        for (int i = 0; i < 2; ++i) {
            int chunk = i * 128 + tid;          // 0..255
            int r = chunk >> 2, k8 = (chunk & 3) * 8;
            async_cp16((uint32_t)(uintptr_t)&As[r * 40 + k8],
                       xh + (size_t)(gm0 + r) * CC + k0 + k8);
            async_cp16((uint32_t)(uintptr_t)&Bs[r * 40 + k8],
                       Wt + (size_t)(gn0 + r) * CC + k0 + k8);
        }
        wait_async0();
        __syncthreads();

        const int row = wv * 16 + lq;
        v16h a = ld16(&As[row * 40 + 8 * hf], &As[row * 40 + 16 + 8 * hf]);
#pragma unroll
        for (int c = 0; c < 4; ++c) {
            int n = c * 16 + lq;
            v16h bf = ld16(&Bs[n * 40 + 8 * hf], &Bs[n * 40 + 16 + 8 * hf]);
            acc[c] = __builtin_amdgcn_wmma_f32_16x16x32_f16(
                false, a, false, bf, (short)0, acc[c], false, false);
        }
    }

    // Epilogue: q,k -> [b][h][n][d]; v -> [b][h][d][n] (transposed)
    const int which = ntile >> 4;   // 64-col tile never straddles q/k/v or head
    const int h     = ntile & 15;
#pragma unroll
    for (int c = 0; c < 4; ++c) {
#pragma unroll
        for (int r = 0; r < 8; ++r) {
            int gm = gm0 + wv * 16 + r + 8 * hf;
            int b = gm >> 11, n = gm & (NN - 1);
            int d = c * 16 + lq;
            size_t off;
            if (which < 2)
                off = (size_t)which * QK + (((size_t)(b * HH + h) * NN + n) * HD + d);
            else
                off = 2 * QK + (((size_t)(b * HH + h) * HD + d) * NN + n);
            qkv[off] = (_Float16)acc[c][r];
        }
    }
}

// ---------------------------------------------------------------------------
// Kernel 2: RoPE on q and k, in place (f16 storage, fp32 math).
// ---------------------------------------------------------------------------
__global__ __launch_bounds__(256) void rope_kernel(
    _Float16* __restrict__ qkv, const float* __restrict__ fc,
    const float* __restrict__ fs) {
    int p = blockIdx.x * 256 + threadIdx.x;   // 2^21 pairs
    int j = p & 31;
    int n = (p >> 5) & (NN - 1);
    int h = (p >> 16) & 15;
    int b = (p >> 20) & 1;
    float cv = fc[((size_t)b * NN + n) * (HD / 2) + j];
    float sv = fs[((size_t)b * NN + n) * (HD / 2) + j];
#pragma unroll
    for (int which = 0; which < 2; ++which) {
        _Float16* ptr = qkv + (size_t)which * QK +
                        (((size_t)(b * HH + h) * NN + n) * HD + 2 * j);
        float t0 = (float)ptr[0], t1 = (float)ptr[1];
        ptr[0] = (_Float16)(t0 * cv - t1 * sv);
        ptr[1] = (_Float16)(t0 * sv + t1 * cv);
    }
}

// ---------------------------------------------------------------------------
// Kernel 3: flash attention. grid=(N/64, B*H), 128 threads (4 waves), and
// ZERO barriers: V is pre-transposed in global so every WMMA fragment is a
// contiguous 16B load; the probs C->A relayout bounces through wave-private
// LDS (same-wave DS ops are in-order per ISA).
// ---------------------------------------------------------------------------
__global__ __launch_bounds__(128) void attn_kernel(
    const _Float16* __restrict__ qkv, float* __restrict__ out) {
    __shared__ alignas(16) _Float16 Ps[4 * 16 * 72];   // per-wave probs 16x64

    const int tid  = threadIdx.x;
    const int wv   = tid >> 5;
    const int lane = tid & 31;
    const int hf   = lane >> 4;
    const int lq   = lane & 15;

    const int qtile = blockIdx.x;          // 0..31
    const int bh    = blockIdx.y;          // 0..31
    const int b = bh >> 4, h = bh & 15;

    const _Float16* qb = qkv + 0 * QK + ((size_t)(b * HH + h) * NN) * HD;
    const _Float16* kb = qkv + 1 * QK + ((size_t)(b * HH + h) * NN) * HD;
    const _Float16* vb = qkv + 2 * QK + ((size_t)(b * HH + h) * HD) * NN;  // [d][n]

    // Q fragments held in registers for the whole kernel
    const int qrow = qtile * 64 + wv * 16 + lq;
    v16h aq[2];
#pragma unroll
    for (int kc = 0; kc < 2; ++kc)
        aq[kc] = ld16(qb + (size_t)qrow * HD + kc * 32 + 8 * hf,
                      qb + (size_t)qrow * HD + kc * 32 + 16 + 8 * hf);

    float m_i[8], l_i[8];
    v8f O[4];
#pragma unroll
    for (int r = 0; r < 8; ++r) { m_i[r] = -3.0e38f; l_i[r] = 0.0f; }
#pragma unroll
    for (int c = 0; c < 4; ++c) O[c] = {};
    const float scale = 0.125f;   // HD^-0.5

    for (int t = 0; t < 32; ++t) {
        // S = Q @ K^T  (16x64 per wave)
        v8f s[4];
#pragma unroll
        for (int c = 0; c < 4; ++c) {
            s[c] = {};
            int krow = t * 64 + c * 16 + lq;
#pragma unroll
            for (int kc = 0; kc < 2; ++kc) {
                v16h bk = ld16(kb + (size_t)krow * HD + kc * 32 + 8 * hf,
                               kb + (size_t)krow * HD + kc * 32 + 16 + 8 * hf);
                s[c] = __builtin_amdgcn_wmma_f32_16x16x32_f16(
                    false, aq[kc], false, bk, (short)0, s[c], false, false);
            }
        }

        // Online softmax per row (row r+8*hf lives in this half's 16 lanes)
#pragma unroll
        for (int r = 0; r < 8; ++r) {
            float s0 = s[0][r] * scale, s1 = s[1][r] * scale;
            float s2 = s[2][r] * scale, s3 = s[3][r] * scale;
            float mx = fmaxf(fmaxf(s0, s1), fmaxf(s2, s3));
#pragma unroll
            for (int off = 1; off < 16; off <<= 1)
                mx = fmaxf(mx, __shfl_xor(mx, off, 32));
            float mn = fmaxf(m_i[r], mx);
            float al = __expf(m_i[r] - mn);
            m_i[r] = mn;
            float p0 = __expf(s0 - mn), p1 = __expf(s1 - mn);
            float p2 = __expf(s2 - mn), p3 = __expf(s3 - mn);
            float rs = p0 + p1 + p2 + p3;
#pragma unroll
            for (int off = 1; off < 16; off <<= 1)
                rs += __shfl_xor(rs, off, 32);
            l_i[r] = l_i[r] * al + rs;
#pragma unroll
            for (int c = 0; c < 4; ++c) O[c][r] = O[c][r] * al;
            int prow = wv * 1152 + (r + 8 * hf) * 72;
            Ps[prow + 0 * 16 + lq] = (_Float16)p0;
            Ps[prow + 1 * 16 + lq] = (_Float16)p1;
            Ps[prow + 2 * 16 + lq] = (_Float16)p2;
            Ps[prow + 3 * 16 + lq] = (_Float16)p3;
        }

        // O += P @ V  (V fragments straight from transposed global layout)
        v16h ap[2];
#pragma unroll
        for (int kc = 0; kc < 2; ++kc)
            ap[kc] = ld16(&Ps[wv * 1152 + lq * 72 + kc * 32 + 8 * hf],
                          &Ps[wv * 1152 + lq * 72 + kc * 32 + 16 + 8 * hf]);
#pragma unroll
        for (int c = 0; c < 4; ++c) {
            const _Float16* vcol = vb + (size_t)(c * 16 + lq) * NN + t * 64;
#pragma unroll
            for (int kc = 0; kc < 2; ++kc) {
                v16h bv = ld16(vcol + kc * 32 + 8 * hf,
                               vcol + kc * 32 + 16 + 8 * hf);
                O[c] = __builtin_amdgcn_wmma_f32_16x16x32_f16(
                    false, ap[kc], false, bv, (short)0, O[c], false, false);
            }
        }
    }

    // y[b][n][h*64+d] = O / l
#pragma unroll
    for (int c = 0; c < 4; ++c) {
#pragma unroll
        for (int r = 0; r < 8; ++r) {
            int n = qtile * 64 + wv * 16 + r + 8 * hf;
            out[((size_t)b * NN + n) * CC + h * HD + c * 16 + lq] = O[c][r] / l_i[r];
        }
    }
}

extern "C" void kernel_launch(void* const* d_in, const int* in_sizes, int n_in,
                              void* d_out, int out_size, void* d_ws, size_t ws_size,
                              hipStream_t stream) {
    (void)in_sizes; (void)n_in; (void)out_size; (void)ws_size;
    const float* x  = (const float*)d_in[0];
    const float* W  = (const float*)d_in[1];
    const float* fc = (const float*)d_in[2];
    const float* fs = (const float*)d_in[3];
    // d_in[4] (attn_mask) is all-true by construction; unused.

    _Float16* qkv = (_Float16*)d_ws;                 // 3*QK f16      (24 MB)
    _Float16* xh  = qkv + 3 * QK;                    // 4096*1024 f16 ( 8 MB)
    _Float16* Wt  = xh + (size_t)4096 * CC;          // 3072*1024 f16 ( 6 MB)
    float* out = (float*)d_out;

    x_convert_kernel<<<16384, 256, 0, stream>>>(x, xh);

    dim3 gt(96, 32);                   // 3072/32 x 1024/32
    w_transpose_kernel<<<gt, 256, 0, stream>>>(W, Wt);

    dim3 g1(48, 64);                   // 3072/64 x 4096/64
    qkv_gemm_kernel<<<g1, 128, 0, stream>>>(xh, Wt, qkv);

    rope_kernel<<<8192, 256, 0, stream>>>(qkv, fc, fs);

    dim3 g3(32, 32);                   // N/64 q-tiles x B*H heads
    attn_kernel<<<g3, 128, 0, stream>>>(qkv, out);
}
